// MHAwithPosEmb_84559316123835
// MI455X (gfx1250) — matrix operand order repaired and checked by reading, not compiled, not run on hardware
//
#include <hip/hip_runtime.h>

// ---------------------------------------------------------------------------
// MHA (B=4, L=S=1024, D=1024, H=16, HD=64) for MI455X / gfx1250.
//
// Roofline: ~43 GFLOP over ~80MB resident (fits in 192MB L2) -> compute
// bound by ~10x at 23.3 TB/s.  All GEMM work runs on the WMMA matrix pipe
// in bf16 with f32 accumulation (v_wmma_f32_16x16x32_bf16); attention is
// flash-style (no 256MB score tensor).  Projection GEMMs stage the weight
// tile into LDS with the Tensor Data Mover (TENSOR_LOAD_TO_LDS), double-
// buffered and synchronized with s_wait_tensorcnt, so DMA overlaps WMMA.
//
// Workspace layout (needs 64 MiB of d_ws):
//   [ 0MB) Xq bf16   [ 8MB) Xk bf16   [16MB) Xv bf16
//   [24MB) Wq bf16   [26MB) Wk bf16   [28MB) Wv bf16   [30MB) Wo bf16
//   [32MB) Qproj     [40MB) Kproj     [48MB) Vproj     [56MB) AttnOut
// ---------------------------------------------------------------------------

typedef __bf16 bf16_t;
typedef __attribute__((ext_vector_type(8)))  __bf16 v8bf;
typedef __attribute__((ext_vector_type(16))) __bf16 v16bf;
typedef __attribute__((ext_vector_type(8)))  float  v8f;
typedef __attribute__((ext_vector_type(4)))  unsigned int v4u;
typedef __attribute__((ext_vector_type(4)))  int v4i;
typedef __attribute__((ext_vector_type(8)))  int v8i;

#define BB   4
#define LL   1024
#define SS   1024
#define DD   1024
#define HH   16
#define HDIM 64

static __device__ __forceinline__ v16bf cat16(v8bf lo, v8bf hi) {
  return __builtin_shufflevector(lo, hi, 0,1,2,3,4,5,6,7,8,9,10,11,12,13,14,15);
}

// WMMA wrapper: D = A(16x32 bf16) * B(32x16 bf16) + C(16x16 f32)
static __device__ __forceinline__ v8f wmma_bf16(v16bf a, v16bf b, v8f c) {
  return __builtin_amdgcn_wmma_f32_16x16x32_bf16(false, a, false, b,
                                                 (short)0, c, false, false);
}

// A-fragment (16x32 bf16, row-major source). rowPtr points at [row][k0] for
// this lane's row (row = base + (lane&15)). Layout (matches B fills):
//   lanes 0-15 : K 0..15
//   lanes 16-31: K 16..31
static __device__ __forceinline__ v16bf load_afrag(const bf16_t* rowPtr, int lane) {
  const int h = (lane >> 4) * 16; // 0 or 16
  v8bf lo = *(const v8bf*)(rowPtr + h);
  v8bf hi = *(const v8bf*)(rowPtr + h + 8);
  return cat16(lo, hi);
}

// ---------------------------------------------------------------------------
// Tensor Data Mover: DMA a 2D tile (tile_x elems contiguous, tile_y rows,
// row stride stride_x elems, 2-byte elements) from global memory into LDS.
// Builds the D# descriptor per CDNA5 ISA 8.3/8.4 (group0 + group1).
// Whole-wave op (EXEC ignored); tracked by TENSORcnt.
// ---------------------------------------------------------------------------
static __device__ __forceinline__ void tdm_load_tile_2d(
    unsigned lds_byte_addr, const bf16_t* gptr,
    unsigned tile_x, unsigned tile_y,
    unsigned tensor_x, unsigned tensor_y, unsigned stride_x) {
  const unsigned long long ga = (unsigned long long)(uintptr_t)gptr;
  v4u g0;
  g0.x = 1u;                                          // count=1, user desc
  g0.y = lds_byte_addr;                               // LDS byte address
  g0.z = (unsigned)(ga & 0xffffffffu);                // global_addr[31:0]
  g0.w = (unsigned)((ga >> 32) & 0x01ffffffu)         // global_addr[56:32]
         | (2u << 30);                                // type = 2 ("image")
  v8i g1;
  g1[0] = (int)(1u << 16);                            // data_size=1 (2B), mask=0
  g1[1] = (int)((tensor_x & 0xffffu) << 16);          // tensor_dim0[15:0]
  g1[2] = (int)(((tensor_x >> 16) & 0xffffu)          // tensor_dim0[31:16]
                | ((tensor_y & 0xffffu) << 16));      // tensor_dim1[15:0]
  g1[3] = (int)(((tensor_y >> 16) & 0xffffu)          // tensor_dim1[31:16]
                | ((tile_x & 0xffffu) << 16));        // tile_dim0
  g1[4] = (int)(tile_y & 0xffffu);                    // tile_dim1 (tile_dim2=0)
  g1[5] = (int)stride_x;                              // tensor_dim0_stride[31:0]
  g1[6] = 0;                                          // stride0 hi / stride1 lo
  g1[7] = 0;                                          // stride1 hi
  v4i z4 = {0, 0, 0, 0};
#if __clang_major__ >= 23
  v8i z8 = {0, 0, 0, 0, 0, 0, 0, 0};                  // amdgpu-toolchain arity
  __builtin_amdgcn_tensor_load_to_lds(g0, g1, z4, z4, z8, 0);
#else
  __builtin_amdgcn_tensor_load_to_lds(g0, g1, z4, z4, 0);   // ROCm 7.2 arity
#endif
}

// ---------------------------------------------------------------------------
// fp32 -> bf16 (round-to-nearest-even via hardware convert)
// ---------------------------------------------------------------------------
__global__ void cvt_f32_bf16(const float* __restrict__ in,
                             bf16_t* __restrict__ out, int n) {
  for (int i = blockIdx.x * blockDim.x + threadIdx.x; i < n;
       i += gridDim.x * blockDim.x)
    out[i] = (bf16_t)in[i];
}

// ---------------------------------------------------------------------------
// C[M,N] = A[M,K] @ W[N,K]^T + bias   (nn.Linear semantics)
// 8 waves / block; block tile 128x64; K step 32.
// The WMMA "B" operand tile (B[k][n] = W[n][k], i.e. 64 rows x 32 elems of
// row-major W) is DMA'd into LDS by the Tensor Data Mover, double-buffered:
// wave 0 issues the next tile's TENSOR_LOAD_TO_LDS before waiting on the
// current one (s_wait_tensorcnt 1 -> overlap DMA with the 4-WMMA compute).
// ---------------------------------------------------------------------------
template <bool OUT_F32>
__global__ __launch_bounds__(256) void gemm_bias(const bf16_t* __restrict__ A,
                                                 const bf16_t* __restrict__ W,
                                                 const float* __restrict__ bias,
                                                 void* __restrict__ outp,
                                                 int M, int N, int K) {
  __shared__ __align__(128) bf16_t bT[2][64 * 32]; // [buf][ncol][klocal]

  const int tid  = threadIdx.x;
  const int lane = tid & 31;
  const int wave = tid >> 5;
  const int n0   = blockIdx.x * 64;
  const int m0   = blockIdx.y * 128;
  const int rowA = m0 + wave * 16 + (lane & 15);
  const int nl   = lane & 15;
  const int kh   = (lane >> 4) * 16; // B-layout: low lanes K 0-15, high K 16-31

  const unsigned ldsB0 =
      (unsigned)(unsigned long long)(uintptr_t)&bT[0][0];
  const unsigned ldsB1 =
      (unsigned)(unsigned long long)(uintptr_t)&bT[1][0];

  v8f acc0 = {}, acc1 = {}, acc2 = {}, acc3 = {};

  // Prologue: DMA first B tile into buffer 0.
  if (wave == 0)
    tdm_load_tile_2d(ldsB0, &W[(size_t)n0 * K], 32, 64,
                     (unsigned)K, (unsigned)N, (unsigned)K);

  for (int k0 = 0; k0 < K; k0 += 32) {
    const int cur = (k0 >> 5) & 1;
    __syncthreads(); // all readers of the next-target buffer are done
    if (wave == 0) {
      if (k0 + 32 < K) {
        tdm_load_tile_2d(cur ? ldsB0 : ldsB1,
                         &W[(size_t)n0 * K + k0 + 32], 32, 64,
                         (unsigned)K, (unsigned)N, (unsigned)K);
        __builtin_amdgcn_s_wait_tensorcnt(1); // current tile landed
      } else {
        __builtin_amdgcn_s_wait_tensorcnt(0);
      }
    }
    __syncthreads(); // DMA'd tile visible to all waves

    const bf16_t* bc = &bT[cur][0];
    v16bf a  = load_afrag(&A[(size_t)rowA * K + k0], lane);
    v16bf b0 = *(const v16bf*)&bc[(0 * 16 + nl) * 32 + kh];
    v16bf b1 = *(const v16bf*)&bc[(1 * 16 + nl) * 32 + kh];
    v16bf b2 = *(const v16bf*)&bc[(2 * 16 + nl) * 32 + kh];
    v16bf b3 = *(const v16bf*)&bc[(3 * 16 + nl) * 32 + kh];
    acc0 = wmma_bf16(a, b0, acc0);
    acc1 = wmma_bf16(a, b1, acc1);
    acc2 = wmma_bf16(a, b2, acc2);
    acc3 = wmma_bf16(a, b3, acc3);
  }

  // Epilogue. C layout: vgpr r -> M = r + 8*(lane>=16), N = (lane&15).
  const int rhi = (lane >> 4) * 8;
#pragma unroll
  for (int c = 0; c < 4; ++c) {
    v8f acc = (c == 0) ? acc0 : (c == 1) ? acc1 : (c == 2) ? acc2 : acc3;
    const int col = n0 + c * 16 + nl;
    const float bv = bias[col];
#pragma unroll
    for (int r = 0; r < 8; ++r) {
      const size_t idx = (size_t)(m0 + wave * 16 + rhi + r) * N + col;
      const float val = acc[r] + bv;
      if (OUT_F32) ((float*)outp)[idx] = val;
      else         ((bf16_t*)outp)[idx] = (bf16_t)val;
    }
  }
}

// ---------------------------------------------------------------------------
// Flash attention: one wave handles (batch b, head h, 16 query rows).
// Processes 32 keys per iteration: QK^T via 4 WMMAs, online softmax with
// 16-lane butterfly reductions, P routed C-layout -> A-layout through LDS,
// V transposed into LDS so PV B-fragments are contiguous ds reads.
// ---------------------------------------------------------------------------
__global__ __launch_bounds__(32) void flash_attn(const bf16_t* __restrict__ q,
                                                 const bf16_t* __restrict__ k,
                                                 const bf16_t* __restrict__ v,
                                                 const unsigned char* __restrict__ mask,
                                                 bf16_t* __restrict__ out) {
  __shared__ __align__(64) bf16_t vT[HDIM * 32]; // [hd][token]
  __shared__ __align__(64) bf16_t pT[16 * 32];   // [row][token]

  const int lane = threadIdx.x;
  const int m0   = blockIdx.x * 16;
  const int h    = blockIdx.y;
  const int b    = blockIdx.z;
  const int nl   = lane & 15;
  const int kh   = (lane >> 4) * 16;

  const bf16_t* qBase = q + (size_t)(b * LL + m0) * DD + h * HDIM;
  const bf16_t* kBase = k + (size_t)b * SS * DD + h * HDIM;
  const bf16_t* vBase = v + (size_t)b * SS * DD + h * HDIM;
  const unsigned char* mBase = mask + b * SS;

  // Q A-fragments for hd chunks [0,32) and [32,64) — reused all iterations.
  const bf16_t* qRow = qBase + (size_t)nl * DD;
  v16bf aQ0 = load_afrag(qRow + 0, lane);
  v16bf aQ1 = load_afrag(qRow + 32, lane);

  v8f o0 = {}, o1 = {}, o2 = {}, o3 = {};
  float mrow[8], lrow[8];
#pragma unroll
  for (int r = 0; r < 8; ++r) { mrow[r] = -3.0e38f; lrow[r] = 0.0f; }

  for (int s0 = 0; s0 < SS; s0 += 32) {
    // ---- scores: two 16x16 tiles over 32 keys --------------------------
    v8f sc[2];
#pragma unroll
    for (int t = 0; t < 2; ++t) {
      // B[k][n] = K[token n][hd k]: token rows are contiguous in memory.
      const bf16_t* kRow = kBase + (size_t)(s0 + t * 16 + nl) * DD;
      v16bf bk0 = *(const v16bf*)(kRow + kh);       // hd 0..31
      v16bf bk1 = *(const v16bf*)(kRow + 32 + kh);  // hd 32..63
      v8f a = {};
      a = wmma_bf16(aQ0, bk0, a);
      a = wmma_bf16(aQ1, bk1, a);
      const bool km = mBase[s0 + t * 16 + nl] != 0; // True = masked
#pragma unroll
      for (int r = 0; r < 8; ++r)
        sc[t][r] = km ? -3.0e38f : a[r] * 0.125f;   // 1/sqrt(64)
    }

    // ---- stage V tile transposed: vT[hd][token] ------------------------
#pragma unroll
    for (int i = 0; i < 8; ++i) {
      const int f = i * 256 + lane * 8;
      const int tok = f >> 6, d = f & 63;
      v8bf x = *(const v8bf*)(vBase + (size_t)(s0 + tok) * DD + d);
#pragma unroll
      for (int j = 0; j < 8; ++j) vT[(d + j) * 32 + tok] = x[j];
    }

    // ---- online softmax (xor masks <16 stay inside the C-layout half) --
    float alpha[8];
#pragma unroll
    for (int r = 0; r < 8; ++r) {
      float x = fmaxf(sc[0][r], sc[1][r]);
      x = fmaxf(x, __shfl_xor(x, 1, 32));
      x = fmaxf(x, __shfl_xor(x, 2, 32));
      x = fmaxf(x, __shfl_xor(x, 4, 32));
      x = fmaxf(x, __shfl_xor(x, 8, 32));
      const float mnew = fmaxf(mrow[r], x);
      alpha[r] = __expf(mrow[r] - mnew);
      mrow[r]  = mnew;
      const float p0 = __expf(sc[0][r] - mnew);
      const float p1 = __expf(sc[1][r] - mnew);
      sc[0][r] = p0; sc[1][r] = p1;
      float s = p0 + p1;
      s += __shfl_xor(s, 1, 32);
      s += __shfl_xor(s, 2, 32);
      s += __shfl_xor(s, 4, 32);
      s += __shfl_xor(s, 8, 32);
      lrow[r] = lrow[r] * alpha[r] + s;
    }
#pragma unroll
    for (int r = 0; r < 8; ++r) {
      o0[r] *= alpha[r]; o1[r] *= alpha[r];
      o2[r] *= alpha[r]; o3[r] *= alpha[r];
    }

    // ---- P: C layout -> row-major LDS ---------------------------------
    const int prow = (lane >> 4) * 8;
#pragma unroll
    for (int t = 0; t < 2; ++t)
#pragma unroll
      for (int r = 0; r < 8; ++r)
        pT[(prow + r) * 32 + t * 16 + nl] = (bf16_t)sc[t][r];
    __syncthreads();

    // ---- O += P(16x32) @ V(32x64) -------------------------------------
    v16bf aP  = load_afrag(&pT[nl * 32], lane);
    v16bf bv0 = *(const v16bf*)&vT[(0 * 16 + nl) * 32 + kh];
    v16bf bv1 = *(const v16bf*)&vT[(1 * 16 + nl) * 32 + kh];
    v16bf bv2 = *(const v16bf*)&vT[(2 * 16 + nl) * 32 + kh];
    v16bf bv3 = *(const v16bf*)&vT[(3 * 16 + nl) * 32 + kh];
    o0 = wmma_bf16(aP, bv0, o0);
    o1 = wmma_bf16(aP, bv1, o1);
    o2 = wmma_bf16(aP, bv2, o2);
    o3 = wmma_bf16(aP, bv3, o3);
    __syncthreads();

    if (s0 + 32 < SS) // pull next K tile toward the WGP (global_prefetch_b8)
      __builtin_prefetch(kBase + (size_t)(s0 + 32 + nl) * DD, 0, 1);
  }

  // ---- epilogue: normalize and store bf16 ------------------------------
  const int rhi = (lane >> 4) * 8;
#pragma unroll
  for (int r = 0; r < 8; ++r) {
    const float inv = 1.0f / lrow[r];
    const size_t base = (size_t)(b * LL + m0 + rhi + r) * DD + h * HDIM;
    out[base + 0 * 16 + nl] = (bf16_t)(o0[r] * inv);
    out[base + 1 * 16 + nl] = (bf16_t)(o1[r] * inv);
    out[base + 2 * 16 + nl] = (bf16_t)(o2[r] * inv);
    out[base + 3 * 16 + nl] = (bf16_t)(o3[r] * inv);
  }
}

// ---------------------------------------------------------------------------
extern "C" void kernel_launch(void* const* d_in, const int* in_sizes, int n_in,
                              void* d_out, int out_size, void* d_ws, size_t ws_size,
                              hipStream_t stream) {
  const float* q_f = (const float*)d_in[0];
  const float* k_f = (const float*)d_in[1];
  const float* v_f = (const float*)d_in[2];
  const unsigned char* mask = (const unsigned char*)d_in[3]; // jax bool = 1 byte
  const float* Wq = (const float*)d_in[4];
  const float* bq = (const float*)d_in[5];
  const float* Wk = (const float*)d_in[6];
  const float* bk = (const float*)d_in[7];
  const float* Wv = (const float*)d_in[8];
  const float* bv = (const float*)d_in[9];
  const float* Wo = (const float*)d_in[10];
  const float* bo = (const float*)d_in[11];

  const size_t ACT = (size_t)BB * LL * DD; // 4Mi elements
  const size_t WEL = (size_t)DD * DD;      // 1Mi elements

  char* ws = (char*)d_ws;
  bf16_t* Xq  = (bf16_t*)(ws + 0);
  bf16_t* Xk  = (bf16_t*)(ws + ACT * 2);
  bf16_t* Xv  = (bf16_t*)(ws + ACT * 4);
  bf16_t* Wqb = (bf16_t*)(ws + ACT * 6);
  bf16_t* Wkb = (bf16_t*)(ws + ACT * 6 + WEL * 2);
  bf16_t* Wvb = (bf16_t*)(ws + ACT * 6 + WEL * 4);
  bf16_t* Wob = (bf16_t*)(ws + ACT * 6 + WEL * 6);
  bf16_t* Qp  = (bf16_t*)(ws + ACT * 6 + WEL * 8);
  bf16_t* Kp  = (bf16_t*)(ws + ACT * 8 + WEL * 8);
  bf16_t* Vp  = (bf16_t*)(ws + ACT * 10 + WEL * 8);
  bf16_t* At  = (bf16_t*)(ws + ACT * 12 + WEL * 8); // ends at 64 MiB

  auto cvt = [&](const float* src, bf16_t* dst, size_t n) {
    int blocks = (int)((n + 255) / 256);
    if (blocks > 2048) blocks = 2048;
    cvt_f32_bf16<<<blocks, 256, 0, stream>>>(src, dst, (int)n);
  };
  cvt(q_f, Xq, ACT);
  cvt(k_f, Xk, ACT);
  cvt(v_f, Xv, ACT);
  cvt(Wq, Wqb, WEL);
  cvt(Wk, Wkb, WEL);
  cvt(Wv, Wvb, WEL);
  cvt(Wo, Wob, WEL);

  const int Mrows = BB * LL; // 4096
  dim3 gGrid(DD / 64, Mrows / 128); // (16, 32)

  gemm_bias<false><<<gGrid, 256, 0, stream>>>(Xq, Wqb, bq, Qp, Mrows, DD, DD);
  gemm_bias<false><<<gGrid, 256, 0, stream>>>(Xk, Wkb, bk, Kp, Mrows, DD, DD);
  gemm_bias<false><<<gGrid, 256, 0, stream>>>(Xv, Wvb, bv, Vp, Mrows, DD, DD);

  flash_attn<<<dim3(LL / 16, HH, BB), 32, 0, stream>>>(Qp, Kp, Vp, mask, At);

  gemm_bias<true><<<gGrid, 256, 0, stream>>>(At, Wob, bo, d_out, Mrows, DD, DD);
}
